// ConditionedKoopmanAFTTransition_30305289240686
// MI455X (gfx1250) — matrix operand order repaired and checked by reading, not compiled, not run on hardware
//
#include <hip/hip_runtime.h>
#include <hip/hip_bf16.h>

// ---------------------------------------------------------------------------
// ConditionedKoopmanAFTTransition rollout for MI455X (gfx1250, wave32, WMMA).
//
// Roofline reasoning: per step ~220 MFLOP after caching window projections,
// ~3MB working set (L2-resident, HBM irrelevant at 23.3 TB/s). The rollout is
// serial in T with a batch-global max per step, so we run 2 graph-captured
// kernels per step (projection GEMM + AFT/residual/observation) and keep all
// GEMMs on v_wmma_f32_16x16x32_f16 (f16 in, f32 accumulate), which is the
// high-throughput matrix path on CDNA5.
// ---------------------------------------------------------------------------

#define B_SZ 512
#define D_SZ 256
#define U_SZ 32
#define T_SZ 128
#define A_SZ 128
#define NOBS 25
#define KE   288   // extended K-dim: D + U_DIM (input term folded into GEMM)
#define NE   640   // fused output cols: K(256) | Wq(128) | Wk(128) | Wv(128)
#define EPSV 1e-8f

typedef _Float16 f16;
typedef __attribute__((ext_vector_type(16))) _Float16 v16h;
typedef __attribute__((ext_vector_type(8)))  _Float16 v8h;
typedef __attribute__((ext_vector_type(8)))  float    v8f;

// A-operand (16x32 f16 tile) from LDS, contiguous k order (same as B):
// lane<16:  row=m0+lane,    halves = K[k0..k0+7]    | K[k0+8..k0+15]
// lane>=16: row=m0+lane-16, halves = K[k0+16..+23]  | K[k0+24..+31]
__device__ inline v16h load_a_lds(const f16* base, int lda, int m0, int k0, int lane) {
  const int hs  = lane >> 4;            // half-select
  const int row = m0 + (lane & 15);
  const f16* p = base + row * lda + k0 + hs * 16;
  v8h lo = *(const v8h*)(p);
  v8h hi = *(const v8h*)(p + 8);
  return __builtin_shufflevector(lo, hi, 0,1,2,3,4,5,6,7,8,9,10,11,12,13,14,15);
}

// B-operand (32x16 f16 tile) from row-major [out,in] weights (i.e. W^T GEMM):
// lane<16: col n=n0+lane, K[k0..k0+15]; lane>=16: n=n0+lane-16, K[k0+16..+31]
__device__ inline v16h load_b_glb(const f16* base, int ldb, int n0, int k0, int lane) {
  const int n = n0 + (lane & 15);
  const int k = k0 + ((lane >> 4) << 4);
  return *(const v16h*)(base + n * ldb + k);
}

// ---------------------------------------------------------------------------
// Prologue: f32 -> f16 weight packing (fused/padded) + initial state staging.
// ---------------------------------------------------------------------------
__global__ __launch_bounds__(256)
void prep_kernel(const float* __restrict__ z0, const float* __restrict__ Kw,
                 const float* __restrict__ Lw, const float* __restrict__ Cw,
                 const float* __restrict__ Dmw, const float* __restrict__ Wq,
                 const float* __restrict__ Wk, const float* __restrict__ Wv,
                 const float* __restrict__ Wo,
                 f16* __restrict__ Wext, f16* __restrict__ Wy,
                 f16* __restrict__ Wo16, f16* __restrict__ Xext) {
  int i = blockIdx.x * 256 + threadIdx.x;
  const int nWext = NE * KE;
  const int nWy   = 32 * KE;
  const int nWo   = D_SZ * A_SZ;
  const int nX    = B_SZ * KE;
  if (i < nWext) {
    int row = i / KE, k = i % KE;
    float v;
    if (row < 256)      v = (k < 256) ? Kw[row * 256 + k]         : Lw[row * 32 + (k - 256)];
    else if (row < 384) v = (k < 256) ? Wq[(row - 256) * 256 + k] : 0.0f;
    else if (row < 512) v = (k < 256) ? Wk[(row - 384) * 256 + k] : 0.0f;
    else                v = (k < 256) ? Wv[(row - 512) * 256 + k] : 0.0f;
    Wext[i] = (f16)v;
  } else if ((i -= nWext) < nWy) {
    int row = i / KE, k = i % KE;
    float v = 0.0f;
    if (row < NOBS) v = (k < 256) ? Cw[row * 256 + k] : Dmw[row * 32 + (k - 256)];
    Wy[i] = (f16)v;
  } else if ((i -= nWy) < nWo) {
    Wo16[i] = (f16)Wo[i];
  } else if ((i -= nWo) < nX) {
    int b = i / KE, k = i % KE;
    Xext[i] = (k < 256) ? (f16)z0[b * 256 + k] : (f16)0.0f;
  }
}

// ---------------------------------------------------------------------------
// Step kernel A: fused projection GEMM  [32x288] x [288 -> 640] per block.
// Produces zlin, sigmoid(zWq^T), K/V projections (ring slot) + partial max.
// ---------------------------------------------------------------------------
__global__ __launch_bounds__(256)
void stepA_kernel(const float* __restrict__ U_t, const float* __restrict__ dtp,
                  f16* __restrict__ Xext, const f16* __restrict__ Wext,
                  float* __restrict__ zlin, float* __restrict__ sigQ,
                  float* __restrict__ KP, float* __restrict__ VP,
                  float* __restrict__ partialMax, int slotCur) {
  __shared__ __align__(16) f16 xS[32][296];   // [z | u*dt] rows, f16, padded
  __shared__ float red[256];
  const int tid = threadIdx.x;
  const int r0  = blockIdx.x * 32;
  const float dt = dtp[0];

  // u*dt -> global Xext (for the observation GEMM in stepB) + LDS
  for (int i = tid; i < 32 * U_SZ; i += 256) {
    int r = i >> 5, u = i & 31;
    f16 h = (f16)(U_t[(r0 + r) * U_SZ + u] * dt);
    Xext[(size_t)(r0 + r) * KE + 256 + u] = h;
    xS[r][256 + u] = h;
  }
  // carry z (f16) from global ring into LDS
  for (int i = tid; i < 32 * D_SZ; i += 256) {
    int r = i >> 8, k = i & 255;
    xS[r][k] = Xext[(size_t)(r0 + r) * KE + k];
  }
  __syncthreads();

  const int wave = tid >> 5, lane = tid & 31;
  float lmax = -3.0e38f;
  for (int tile = wave; tile < 2 * 40; tile += 8) {
    int mt = tile / 40, nt = tile % 40;
    int m0 = mt * 16, n0 = nt * 16;
    v8f acc = {};
#pragma unroll
    for (int kt = 0; kt < 9; ++kt) {
      v16h a = load_a_lds(&xS[0][0], 296, m0, kt * 32, lane);
      v16h b = load_b_glb(Wext, KE, n0, kt * 32, lane);
      acc = __builtin_amdgcn_wmma_f32_16x16x32_f16(false, a, false, b,
                                                   (short)0, acc, false, false);
    }
    const int col     = n0 + (lane & 15);
    const int rowbase = m0 + ((lane >> 4) << 3);
#pragma unroll
    for (int r = 0; r < 8; ++r) {
      int grow = r0 + rowbase + r;
      float v = acc[r];
      if (col < 256) {
        zlin[(size_t)grow * D_SZ + col] = v;
      } else if (col < 384) {
        sigQ[(size_t)grow * A_SZ + (col - 256)] = 1.0f / (1.0f + __expf(-v));
      } else if (col < 512) {
        KP[(size_t)slotCur * B_SZ * A_SZ + (size_t)grow * A_SZ + (col - 384)] = v;
        lmax = fmaxf(lmax, v);
      } else {
        VP[(size_t)slotCur * B_SZ * A_SZ + (size_t)grow * A_SZ + (col - 512)] = v;
      }
    }
  }
  // block max of the fresh K-projection (batch-global max finished in stepB)
  red[tid] = lmax;
  __syncthreads();
  for (int s = 128; s > 0; s >>= 1) {
    if (tid < s) red[tid] = fmaxf(red[tid], red[tid + s]);
    __syncthreads();
  }
  if (tid == 0) partialMax[blockIdx.x] = red[0];
}

// ---------------------------------------------------------------------------
// Step kernel B: AFT gating (elementwise), residual GEMM [32x128]x[128->256],
// z_next = zlin + residual, observation GEMM [32x288]x[288->32].
// ---------------------------------------------------------------------------
__global__ __launch_bounds__(256)
void stepB_kernel(f16* __restrict__ Xext, const f16* __restrict__ Wo16,
                  const f16* __restrict__ Wy16, const float* __restrict__ zlin,
                  const float* __restrict__ sigQ, const float* __restrict__ KP,
                  const float* __restrict__ VP, const float* __restrict__ partialMax,
                  float* __restrict__ maxK, float* __restrict__ outZ,
                  float* __restrict__ outY,
                  int slot0, int slot1, int slot2, int mask0i, int mask1i, int t) {
  __shared__ __align__(16) f16 gS[32][136];   // gated AFT output, f16
  __shared__ __align__(16) f16 zS[32][296];   // [z_next | u*dt] for y GEMM
  __shared__ float msh[4];
  const int tid = threadIdx.x;
  const int r0  = blockIdx.x * 32;

  if (tid == 0) {
    float m = -3.0e38f;
    for (int i = 0; i < 16; ++i) m = fmaxf(m, partialMax[i]);
    maxK[slot2] = m;                    // cache for the next two steps
    msh[2] = m;
    msh[0] = mask0i ? maxK[slot0] : 0.0f;
    msh[1] = mask1i ? maxK[slot1] : 0.0f;
  }
  __syncthreads();
  const float m0v = msh[0], m1v = msh[1], m2v = msh[2];

  // AFT: masked exp-weighted average over the 3 cached window projections
  for (int i = tid; i < 32 * A_SZ; i += 256) {
    int r = i >> 7, a = i & 127;
    size_t idx = (size_t)(r0 + r) * A_SZ + a;
    float num = 0.0f, den = EPSV;
    if (mask0i) {
      float e = __expf(KP[(size_t)slot0 * B_SZ * A_SZ + idx] - m0v);
      num += e * VP[(size_t)slot0 * B_SZ * A_SZ + idx]; den += e;
    }
    if (mask1i) {
      float e = __expf(KP[(size_t)slot1 * B_SZ * A_SZ + idx] - m1v);
      num += e * VP[(size_t)slot1 * B_SZ * A_SZ + idx]; den += e;
    }
    {
      float e = __expf(KP[(size_t)slot2 * B_SZ * A_SZ + idx] - m2v);
      num += e * VP[(size_t)slot2 * B_SZ * A_SZ + idx]; den += e;
    }
    gS[r][a] = (f16)(sigQ[idx] * (num / den));
  }
  // u*dt columns for the observation GEMM
  for (int i = tid; i < 32 * U_SZ; i += 256) {
    int r = i >> 5, u = i & 31;
    zS[r][256 + u] = Xext[(size_t)(r0 + r) * KE + 256 + u];
  }
  __syncthreads();

  const int wave = tid >> 5, lane = tid & 31;
  // residual = g @ Wo^T ; z_next = zlin + residual
  for (int tile = wave; tile < 2 * 16; tile += 8) {
    int mt = tile >> 4, nt = tile & 15;
    int m0t = mt * 16, n0 = nt * 16;
    v8f acc = {};
#pragma unroll
    for (int kt = 0; kt < 4; ++kt) {
      v16h a = load_a_lds(&gS[0][0], 136, m0t, kt * 32, lane);
      v16h b = load_b_glb(Wo16, A_SZ, n0, kt * 32, lane);
      acc = __builtin_amdgcn_wmma_f32_16x16x32_f16(false, a, false, b,
                                                   (short)0, acc, false, false);
    }
    const int col  = n0 + (lane & 15);
    const int rowb = m0t + ((lane >> 4) << 3);
#pragma unroll
    for (int r = 0; r < 8; ++r) {
      int lrow = rowb + r;
      int grow = r0 + lrow;
      float zn = acc[r] + zlin[(size_t)grow * D_SZ + col];
      outZ[((size_t)t * B_SZ + grow) * D_SZ + col] = zn;   // Zs[t]
      f16 h = (f16)zn;
      Xext[(size_t)grow * KE + col] = h;                   // carry for step t+1
      zS[lrow][col] = h;                                   // for y GEMM below
    }
  }
  __syncthreads();

  // y = z_next @ C^T + (u*dt) @ Dm^T  via padded [32x288]x[288->32] GEMM
  if (wave < 4) {
    int m0t = (wave >> 1) * 16, n0 = (wave & 1) * 16;
    v8f acc = {};
#pragma unroll
    for (int kt = 0; kt < 9; ++kt) {
      v16h a = load_a_lds(&zS[0][0], 296, m0t, kt * 32, lane);
      v16h b = load_b_glb(Wy16, KE, n0, kt * 32, lane);
      acc = __builtin_amdgcn_wmma_f32_16x16x32_f16(false, a, false, b,
                                                   (short)0, acc, false, false);
    }
    const int col = n0 + (lane & 15);
    if (col < NOBS) {
      const int rowb = m0t + ((lane >> 4) << 3);
#pragma unroll
      for (int r = 0; r < 8; ++r) {
        int grow = r0 + rowb + r;
        outY[(size_t)t * B_SZ * NOBS + (size_t)grow * NOBS + col] = acc[r];
      }
    }
  }
}

// ---------------------------------------------------------------------------
extern "C" void kernel_launch(void* const* d_in, const int* in_sizes, int n_in,
                              void* d_out, int out_size, void* d_ws, size_t ws_size,
                              hipStream_t stream) {
  const float* z0  = (const float*)d_in[0];
  // d_in[1] = z_static (unused, matches reference)
  const float* dtp = (const float*)d_in[2];
  const float* U   = (const float*)d_in[3];
  const float* Kw  = (const float*)d_in[4];
  const float* Lw  = (const float*)d_in[5];
  const float* Cw  = (const float*)d_in[6];
  const float* Dmw = (const float*)d_in[7];
  const float* Wq  = (const float*)d_in[8];
  const float* Wk  = (const float*)d_in[9];
  const float* Wv  = (const float*)d_in[10];
  const float* Wo  = (const float*)d_in[11];

  char* ws = (char*)d_ws;
  auto alloc = [&](size_t bytes) -> char* {
    char* p = ws;
    ws += (bytes + 255) & ~(size_t)255;
    return p;
  };
  f16*   Xext = (f16*)alloc((size_t)B_SZ * KE * sizeof(f16));
  f16*   Wext = (f16*)alloc((size_t)NE * KE * sizeof(f16));
  f16*   Wy16 = (f16*)alloc((size_t)32 * KE * sizeof(f16));
  f16*   Wo16 = (f16*)alloc((size_t)D_SZ * A_SZ * sizeof(f16));
  float* zlin = (float*)alloc((size_t)B_SZ * D_SZ * sizeof(float));
  float* sigQ = (float*)alloc((size_t)B_SZ * A_SZ * sizeof(float));
  float* KP   = (float*)alloc((size_t)3 * B_SZ * A_SZ * sizeof(float));
  float* VP   = (float*)alloc((size_t)3 * B_SZ * A_SZ * sizeof(float));
  float* pmax = (float*)alloc(64 * sizeof(float));
  float* maxK = (float*)alloc(16 * sizeof(float));

  float* outZ = (float*)d_out;
  float* outY = outZ + (size_t)T_SZ * B_SZ * D_SZ;

  const int totalPrep = NE * KE + 32 * KE + D_SZ * A_SZ + B_SZ * KE;
  prep_kernel<<<(totalPrep + 255) / 256, 256, 0, stream>>>(
      z0, Kw, Lw, Cw, Dmw, Wq, Wk, Wv, Wo, Wext, Wy16, Wo16, Xext);

  for (int t = 0; t < T_SZ; ++t) {
    const int s2 = t % 3;         // slot of z_t   (fresh projection)
    const int s1 = (t + 2) % 3;   // slot of z_{t-1}
    const int s0 = (t + 1) % 3;   // slot of z_{t-2}
    stepA_kernel<<<16, 256, 0, stream>>>(U + (size_t)t * B_SZ * U_SZ, dtp,
                                         Xext, Wext, zlin, sigQ, KP, VP, pmax, s2);
    stepB_kernel<<<16, 256, 0, stream>>>(Xext, Wo16, Wy16, zlin, sigQ, KP, VP,
                                         pmax, maxK, outZ, outY, s0, s1, s2,
                                         (t >= 2) ? 1 : 0, (t >= 1) ? 1 : 0, t);
  }
}
